// AttentionHead_73203422593271
// MI455X (gfx1250) — compile-verified, matrix operand order checked by reading, not run on hardware
//
#include <hip/hip_runtime.h>
#include <stdint.h>

#define EMB 32
#define HD  64
#define BB  4
#define SS  4096

typedef __attribute__((ext_vector_type(16))) __bf16 v16bf;
typedef __attribute__((ext_vector_type(8)))  __bf16 v8bf;
typedef __attribute__((ext_vector_type(16))) float  v16f;
typedef __attribute__((ext_vector_type(8)))  float  v8f;
typedef __attribute__((ext_vector_type(4)))  float  v4f;
typedef __attribute__((ext_vector_type(4)))  unsigned uv4;

// native f32 -> bf16 conversion; let the backend pick v_cvt_pk_bf16_f32 etc.
static __device__ __forceinline__ __bf16 f2bf(float f) { return (__bf16)f; }

// Load one 16-half A/B fragment: elems [0..7] from p[0..7], [8..15] from p[16..23].
// ISA 16-bit A/B layout: per-lane K = kbase + (j<8 ? j : j+8).
static __device__ __forceinline__ v16bf load_frag(const __bf16* p) {
    v8bf a = *reinterpret_cast<const v8bf*>(p);
    v8bf b = *reinterpret_cast<const v8bf*>(p + 16);
    v16bf r;
#pragma unroll
    for (int i = 0; i < 8; ++i) { r[i] = a[i]; r[i + 8] = b[i]; }
    return r;
}

// ---------------------------------------------------------------------------
// Kernel 1: QKV projection.  One wave per 16-row tile of x (flat over B*S).
// Q,K row-major bf16 [B*S][64];  V transposed bf16 [B][64][S].
// ---------------------------------------------------------------------------
__global__ __launch_bounds__(256)
void qkv_project_kernel(const float* __restrict__ x,
                        const float* __restrict__ Wq, const float* __restrict__ bq,
                        const float* __restrict__ Wk, const float* __restrict__ bk,
                        const float* __restrict__ Wv, const float* __restrict__ bv,
                        __bf16* __restrict__ Q, __bf16* __restrict__ K,
                        __bf16* __restrict__ Vt)
{
    const int lane = threadIdx.x & 31;
    const int w    = threadIdx.x >> 5;
    const int lp   = lane & 15;
    const int hi   = lane >> 4;
    const int kb   = hi * 8;

    const int rowbase = (blockIdx.x * 8 + w) * 16;     // flat over B*S
    const int b  = rowbase / SS;
    const int sl = rowbase % SS;

    // A-fragment of x tile (16 rows x 32): vector loads, one packed convert
    const float* xr = x + (size_t)(rowbase + lp) * EMB + kb;
    v4f x0 = *reinterpret_cast<const v4f*>(xr);
    v4f x1 = *reinterpret_cast<const v4f*>(xr + 4);
    v4f x2 = *reinterpret_cast<const v4f*>(xr + 16);
    v4f x3 = *reinterpret_cast<const v4f*>(xr + 20);
    v16f xa;
#pragma unroll
    for (int i = 0; i < 4; ++i) {
        xa[i] = x0[i]; xa[4 + i] = x1[i]; xa[8 + i] = x2[i]; xa[12 + i] = x3[i];
    }
    const v16bf a = __builtin_convertvector(xa, v16bf);

    const float* Ws[3]   = {Wq, Wk, Wv};
    const float* bias[3] = {bq, bk, bv};

#pragma unroll
    for (int mtx = 0; mtx < 3; ++mtx) {
        const float* W   = Ws[mtx];
        const float* bv_ = bias[mtx];
#pragma unroll
        for (int nt = 0; nt < 4; ++nt) {
            const int col = nt * 16 + lp;
            v16f wf;
#pragma unroll
            for (int j = 0; j < 8; ++j) {
                wf[j]     = W[(kb + j) * HD + col];
                wf[j + 8] = W[(kb + 16 + j) * HD + col];
            }
            const v16bf bw = __builtin_convertvector(wf, v16bf);

            const float bbias = bv_[col];
            v8f c;
#pragma unroll
            for (int i = 0; i < 8; ++i) c[i] = bbias;  // fold bias into C
            c = __builtin_amdgcn_wmma_f32_16x16x32_bf16(
                    false, a, false, bw, (short)0, c, false, false);

            if (mtx == 0) {
#pragma unroll
                for (int r = 0; r < 8; ++r)
                    Q[(size_t)(rowbase + r + 8 * hi) * HD + col] = f2bf(c[r]);
            } else if (mtx == 1) {
#pragma unroll
                for (int r = 0; r < 8; ++r)
                    K[(size_t)(rowbase + r + 8 * hi) * HD + col] = f2bf(c[r]);
            } else {
                // Vt[b][col][sl+8hi+r]: packed convert + single b128 store
                v8bf cb = __builtin_convertvector(c, v8bf);
                *reinterpret_cast<uv4*>(Vt + (size_t)b * HD * SS +
                                        (size_t)col * SS + sl + 8 * hi) =
                    __builtin_bit_cast(uv4, cb);
            }
        }
    }
}

// ---------------------------------------------------------------------------
// Kernel 2: causal flash attention, transposed-score formulation.
//   S^T = K * Q^T   (D lane = query column -> lane-local softmax)
//   O^T = V^T * P^T (D lane = query column -> lane-local rescale, vector stores)
// No LDS, 2 shuffles per 32-key chunk.
// ---------------------------------------------------------------------------
__global__ __launch_bounds__(256)
void flash_attn_kernel(const __bf16* __restrict__ Q, const __bf16* __restrict__ K,
                       const __bf16* __restrict__ Vt, float* __restrict__ out)
{
    const int lane = threadIdx.x & 31;
    const int w    = threadIdx.x >> 5;
    const int lp   = lane & 15;
    const int hi   = lane >> 4;
    const int kb   = hi * 8;

    const int blocksPerBatch = SS / 128;               // 32
    const int batch = blockIdx.x / blocksPerBatch;
    const int qbase = (blockIdx.x % blocksPerBatch) * 128 + w * 16;
    const int query = qbase + lp;                      // this lane's query row

    const float SC2     = 0.125f * 1.44269504088896340736f;  // scale * log2(e)
    const float NEG_INF = -__builtin_inff();

    // Q as B-matrix (Q^T): B-frag lane col = query, elems = h (kb pattern)
    const __bf16* qrow = Q + (size_t)(batch * SS + query) * HD;
    const v16bf qf0 = load_frag(qrow + kb);
    const v16bf qf1 = load_frag(qrow + 32 + kb);

    float m = NEG_INF, l = 0.0f;
    v8f acc[4];
#pragma unroll
    for (int t = 0; t < 4; ++t)
#pragma unroll
        for (int r = 0; r < 8; ++r) acc[t][r] = 0.0f;

    const int nch = (qbase >> 5) + 1;                  // causal 32-key chunks
    for (int ch = 0; ch < nch; ++ch) {
        const int kvb = ch * 32;

        // ---- S^T tiles: A = K rows (keys), B = Q^T ----
        v8f s0, s1;
#pragma unroll
        for (int nt = 0; nt < 2; ++nt) {
            const __bf16* krow = K + (size_t)(batch * SS + kvb + nt * 16 + lp) * HD;
            v16bf kf0 = load_frag(krow + kb);
            v16bf kf1 = load_frag(krow + 32 + kb);
            v8f c;
#pragma unroll
            for (int i = 0; i < 8; ++i) c[i] = 0.0f;
            c = __builtin_amdgcn_wmma_f32_16x16x32_bf16(
                    false, kf0, false, qf0, (short)0, c, false, false);
            c = __builtin_amdgcn_wmma_f32_16x16x32_bf16(
                    false, kf1, false, qf1, (short)0, c, false, false);
            if (nt == 0) s0 = c; else s1 = c;
        }

        // ---- lane-local masked online softmax (base-2 domain) ----
        float sv0[8], sv1[8];
#pragma unroll
        for (int r = 0; r < 8; ++r) {
            const int k0 = kvb + r + 8 * hi;           // key of s0[r]
            const int k1 = kvb + 16 + r + 8 * hi;      // key of s1[r]
            const float r0 = s0[r], r1 = s1[r];
            // reference semantics: tril + (exact zero -> -inf)
            sv0[r] = (k0 <= query && r0 != 0.0f) ? r0 * SC2 : NEG_INF;
            sv1[r] = (k1 <= query && r1 != 0.0f) ? r1 * SC2 : NEG_INF;
        }
        float mx = NEG_INF;
#pragma unroll
        for (int r = 0; r < 8; ++r) { mx = fmaxf(mx, sv0[r]); mx = fmaxf(mx, sv1[r]); }
        mx = fmaxf(mx, __shfl_xor(mx, 16, 32));        // combine key halves

        const float mn    = fmaxf(m, mx);
        const bool  dead  = (mn == NEG_INF);
        const float mnS   = dead ? 0.0f : mn;
        const float aArg  = dead ? 0.0f : (m - mn);
        const float alpha = exp2f(aArg);

        v16f pv;
        float rs = 0.0f;
#pragma unroll
        for (int r = 0; r < 8; ++r) {
            const float e0 = exp2f(sv0[r] - mnS);
            const float e1 = exp2f(sv1[r] - mnS);
            pv[r] = e0; pv[8 + r] = e1;                // elem j<8 <- s0, else s1
            rs += e0 + e1;
        }
        rs += __shfl_xor(rs, 16, 32);                  // combine key halves

        m = mn;
        l = l * alpha + rs;
#pragma unroll
        for (int t = 0; t < 4; ++t)
#pragma unroll
            for (int r = 0; r < 8; ++r) acc[t][r] *= alpha;

        // packed convert: P fragment already in A/B per-lane layout
        const v16bf pf = __builtin_convertvector(pv, v16bf);

        // ---- O^T += V^T * P^T : A = Vt fragment, B = P fragment ----
#pragma unroll
        for (int ht = 0; ht < 4; ++ht) {
            const __bf16* vrow = Vt + (size_t)batch * HD * SS +
                                 (size_t)(ht * 16 + lp) * SS + kvb + kb;
            v16bf vf = load_frag(vrow);
            acc[ht] = __builtin_amdgcn_wmma_f32_16x16x32_bf16(
                          false, vf, false, pf, (short)0, acc[ht], false, false);
        }
    }

    // ---- epilogue: lane-local normalize, contiguous vector stores ----
    const float invl = 1.0f / l;
#pragma unroll
    for (int ht = 0; ht < 4; ++ht) {
        float* op = out + (size_t)(batch * SS + query) * HD + ht * 16 + 8 * hi;
        v4f lo, hiV;
#pragma unroll
        for (int i = 0; i < 4; ++i) { lo[i] = acc[ht][i] * invl; hiV[i] = acc[ht][4 + i] * invl; }
        *reinterpret_cast<v4f*>(op)     = lo;
        *reinterpret_cast<v4f*>(op + 4) = hiV;
    }
}

// ---------------------------------------------------------------------------
extern "C" void kernel_launch(void* const* d_in, const int* in_sizes, int n_in,
                              void* d_out, int out_size, void* d_ws, size_t ws_size,
                              hipStream_t stream) {
    const float* x  = (const float*)d_in[0];
    const float* Wq = (const float*)d_in[1];
    const float* bq = (const float*)d_in[2];
    const float* Wk = (const float*)d_in[3];
    const float* bk = (const float*)d_in[4];
    const float* Wv = (const float*)d_in[5];
    const float* bv = (const float*)d_in[6];

    const size_t nElem = (size_t)BB * SS * HD;          // 1M bf16 per tensor
    __bf16* qws = (__bf16*)d_ws;
    __bf16* kws = qws + nElem;
    __bf16* vws = kws + nElem;                          // 6 MB total

    const int blocks = (BB * SS) / 128;                 // 128 blocks, 8 waves each
    qkv_project_kernel<<<blocks, 256, 0, stream>>>(x, Wq, bq, Wk, bk, Wv, bv,
                                                   qws, kws, vws);
    flash_attn_kernel<<<blocks, 256, 0, stream>>>(qws, kws, vws, (float*)d_out);
}
